// Decoder_8340826489417
// MI455X (gfx1250) — matrix operand order repaired and checked by reading, not compiled, not run on hardware
//
#include <hip/hip_runtime.h>
#include <hip/hip_bf16.h>

typedef __attribute__((ext_vector_type(16))) _Float16 v16h;
typedef __attribute__((ext_vector_type(8)))  _Float16 v8h;
typedef __attribute__((ext_vector_type(8)))  float    v8f;

#define SEQ_LEN 12
#define H_DIM   128
#define EMB     64
#define KTOT    192          // EMB + H_DIM
#define G4      512          // 4 * H_DIM
#define NPEDS   65536
#define WAVES   8
#define TPB     (WAVES * 32)
#define PEDS_PER_WAVE  16
#define PEDS_PER_BLOCK (WAVES * PEDS_PER_WAVE)

// ---- LDS layout (dynamic shared) ----
// Wsh   : [512][192] f16   = 196608 B   (combined [w_ih | w_hh], row = gate, col = K)
// Xsh   : [8][16][192] f16 =  49152 B   (per-wave activation tile: [dec_in(64) | h(128)])
// bias  : [512] f32        =   2048 B   (b_ih + b_hh)
// whp   : [2][128] f32     =   1024 B
// bhp   : [2] f32          =      8 B
// wse   : [64][2] f32      =    512 B
// bse   : [64] f32         =    256 B
// relbuf: [8][2][16] f32   =   1024 B
#define SMEM_BYTES (196608 + 49152 + 2048 + 1024 + 8 + 512 + 256 + 1024)

// Fast single-instruction v_rcp_f32: ~1 ulp, negligible next to the f16
// quantization of the matmul path; avoids clang's precise-div expansion.
__device__ __forceinline__ float fast_rcp(float x) { return __builtin_amdgcn_rcpf(x); }

__device__ __forceinline__ float sigmoid_f(float x) {
    return fast_rcp(1.0f + __expf(-x));
}
__device__ __forceinline__ float tanh_f(float x) {
    // overflow-safe tanh via exp(-2|x|)
    float t = __expf(-2.0f * fabsf(x));
    float r = (1.0f - t) * fast_rcp(1.0f + t);
    return copysignf(r, x);
}

// A-fragment (16x32 f16, contiguous k order, same as B): row-major [m][K] LDS source.
// lanes 0-15 : M = lane,   K = kBase+0..15 (contiguous)
// lanes 16-31: M = lane-16, K = kBase+16..31
__device__ __forceinline__ v16h load_fragA(const _Float16* base, int kBase, int lane) {
    int hi = lane >> 4;
    int m  = lane & 15;
    const _Float16* p = base + m * KTOT + kBase + hi * 16;
    v8h lo = *(const v8h*)(p);       // K +0..7
    v8h hh = *(const v8h*)(p + 8);   // K +8..15
    v16h r;
#pragma unroll
    for (int i = 0; i < 8; ++i) { r[i] = lo[i]; r[i + 8] = hh[i]; }
    return r;
}

// B-fragment (32x16 f16, same contiguous k order as A): W stored row-major [n][K].
// lanes 0-15 : N = lane,    K = kBase+0..15 (contiguous)
// lanes 16-31: N = lane-16, K = kBase+16..31
__device__ __forceinline__ v16h load_fragB(const _Float16* Wsh, int n, int kBase, int hi) {
    return *(const v16h*)(Wsh + n * KTOT + kBase + hi * 16);
}

__global__ __launch_bounds__(TPB)
void decoder_lstm_wmma(const float* __restrict__ last_pos_rel,
                       const float* __restrict__ h0,
                       const float* __restrict__ c0,
                       const float* __restrict__ w_ih, const float* __restrict__ w_hh,
                       const float* __restrict__ b_ih, const float* __restrict__ b_hh,
                       const float* __restrict__ w_se, const float* __restrict__ b_se,
                       const float* __restrict__ w_hp, const float* __restrict__ b_hp,
                       float* __restrict__ out)
{
    extern __shared__ char smem[];
    _Float16* Wsh   = (_Float16*)smem;                         // [512][192]
    _Float16* Xsh   = (_Float16*)(smem + 196608);              // [8][16][192]
    float*    biass = (float*)(smem + 196608 + 49152);         // [512]
    float*    whps  = biass + G4;                              // [2][128]
    float*    bhps  = whps + 2 * H_DIM;                        // [2]
    float*    wses  = bhps + 2;                                // [64][2]
    float*    bses  = wses + 2 * EMB;                          // [64]
    float*    relb  = bses + EMB;                              // [8][2][16]

    const int tid = threadIdx.x;

    // ---- warm L2 for the one-time f32 weight fetch (global_prefetch_b8) ----
    for (int idx = tid * 64; idx < G4 * EMB; idx += TPB * 64)
        __builtin_prefetch(&w_ih[idx], 0, 0);
    for (int idx = tid * 64; idx < G4 * H_DIM; idx += TPB * 64)
        __builtin_prefetch(&w_hh[idx], 0, 0);

    // ---- cooperative weight staging (one-time) ----
    for (int idx = tid; idx < G4 * KTOT; idx += TPB) {
        int n = idx / KTOT, k = idx - n * KTOT;
        float v = (k < EMB) ? w_ih[n * EMB + k] : w_hh[n * H_DIM + (k - EMB)];
        Wsh[idx] = (_Float16)v;
    }
    for (int idx = tid; idx < G4; idx += TPB) biass[idx] = b_ih[idx] + b_hh[idx];
    for (int idx = tid; idx < 2 * H_DIM; idx += TPB) whps[idx] = w_hp[idx];
    if (tid < 2) bhps[tid] = b_hp[tid];
    for (int idx = tid; idx < 2 * EMB; idx += TPB) wses[idx] = w_se[idx];
    for (int idx = tid; idx < EMB; idx += TPB) bses[idx] = b_se[idx];
    __syncthreads();

    const int w    = tid >> 5;
    const int lane = tid & 31;
    const int hi   = lane >> 4;
    const int nl   = lane & 15;
    const int pedBase = blockIdx.x * PEDS_PER_BLOCK + w * PEDS_PER_WAVE;
    _Float16* X  = Xsh + w * (PEDS_PER_WAVE * KTOT);
    float*    rb = relb + w * 32;

    // ---- per-wave state init ----
    // c0 into registers, WMMA C-layout: VGPR j -> row m = j + hi*8, col = lane&15
    v8f c_t[8];
#pragma unroll
    for (int ht = 0; ht < 8; ++ht) {
#pragma unroll
        for (int j = 0; j < 8; ++j) {
            int m = j + hi * 8;
            c_t[ht][j] = c0[(size_t)(pedBase + m) * H_DIM + ht * 16 + nl];
        }
    }
    // h0 -> X[:, 64:192] as f16 (lane: row nl, k-half hi)
    {
#pragma unroll 4
        for (int k = 0; k < 64; ++k) {
            int kk = hi * 64 + k;
            X[nl * KTOT + EMB + kk] = (_Float16)h0[(size_t)(pedBase + nl) * H_DIM + kk];
        }
    }
    // dec_in0 = last_pos_rel @ w_se.T + b_se -> X[:, 0:64]
    {
        float r0 = last_pos_rel[(size_t)(pedBase + nl) * 2 + 0];
        float r1 = last_pos_rel[(size_t)(pedBase + nl) * 2 + 1];
#pragma unroll 4
        for (int j = 0; j < 32; ++j) {
            int e = hi * 32 + j;
            X[nl * KTOT + e] = (_Float16)(r0 * wses[e * 2 + 0] + r1 * wses[e * 2 + 1] + bses[e]);
        }
    }

    // ---- 12-step recurrence (wave-private; no cross-wave sync needed) ----
#pragma unroll 1
    for (int t = 0; t < SEQ_LEN; ++t) {
        // A fragments for X[16 x 192]: 6 k-tiles of 32
        v16h A[6];
#pragma unroll
        for (int kt = 0; kt < 6; ++kt) A[kt] = load_fragA(X, kt * 32, lane);

#pragma unroll
        for (int ht = 0; ht < 8; ++ht) {
            v8f gi = {}, gf = {}, gg = {}, go = {};
            const int ni = 0 * H_DIM + ht * 16 + nl;
            const int nf = 1 * H_DIM + ht * 16 + nl;
            const int ng = 2 * H_DIM + ht * 16 + nl;
            const int no = 3 * H_DIM + ht * 16 + nl;
#pragma unroll
            for (int kt = 0; kt < 6; ++kt) {
                v16h Bi = load_fragB(Wsh, ni, kt * 32, hi);
                gi = __builtin_amdgcn_wmma_f32_16x16x32_f16(false, A[kt], false, Bi, (short)0, gi, false, false);
                v16h Bf = load_fragB(Wsh, nf, kt * 32, hi);
                gf = __builtin_amdgcn_wmma_f32_16x16x32_f16(false, A[kt], false, Bf, (short)0, gf, false, false);
                v16h Bg = load_fragB(Wsh, ng, kt * 32, hi);
                gg = __builtin_amdgcn_wmma_f32_16x16x32_f16(false, A[kt], false, Bg, (short)0, gg, false, false);
                v16h Bo = load_fragB(Wsh, no, kt * 32, hi);
                go = __builtin_amdgcn_wmma_f32_16x16x32_f16(false, A[kt], false, Bo, (short)0, go, false, false);
            }
            float bi = biass[ni], bf = biass[nf], bg = biass[ng], bo = biass[no];
#pragma unroll
            for (int j = 0; j < 8; ++j) {
                float iv = sigmoid_f(gi[j] + bi);
                float fv = sigmoid_f(gf[j] + bf);
                float gv = tanh_f(gg[j] + bg);
                float ov = sigmoid_f(go[j] + bo);
                float cv = fv * c_t[ht][j] + iv * gv;
                c_t[ht][j] = cv;
                float hv = ov * tanh_f(cv);
                int m = j + hi * 8;
                X[m * KTOT + EMB + ht * 16 + nl] = (_Float16)hv;  // h back into X (f16)
            }
        }

        // rel = h @ w_hp.T + b_hp   (lane: ped nl, output component hi)
        {
            float acc = 0.0f;
            const _Float16* hrow = X + nl * KTOT + EMB;
            const float* wr = whps + hi * H_DIM;
#pragma unroll 8
            for (int k = 0; k < H_DIM; ++k) acc = fmaf((float)hrow[k], wr[k], acc);
            float rel = acc + bhps[hi];
            out[((size_t)t * NPEDS + (pedBase + nl)) * 2 + hi] = rel;
            rb[hi * 16 + nl] = rel;
        }
        // dec_in = rel @ w_se.T + b_se -> X[:, 0:64]
        {
            float r0 = rb[nl], r1 = rb[16 + nl];
#pragma unroll 4
            for (int j = 0; j < 32; ++j) {
                int e = hi * 32 + j;
                X[nl * KTOT + e] = (_Float16)(r0 * wses[e * 2 + 0] + r1 * wses[e * 2 + 1] + bses[e]);
            }
        }
    }
}

extern "C" void kernel_launch(void* const* d_in, const int* in_sizes, int n_in,
                              void* d_out, int out_size, void* d_ws, size_t ws_size,
                              hipStream_t stream) {
    (void)in_sizes; (void)n_in; (void)d_ws; (void)ws_size; (void)out_size;
    // setup_inputs() order:
    // 0 last_pos (unused in output), 1 last_pos_rel, 2 h0, 3 c0,
    // 4 w_ih, 5 w_hh, 6 b_ih, 7 b_hh, 8 w_se, 9 b_se, 10 w_hp, 11 b_hp
    const float* last_pos_rel = (const float*)d_in[1];
    const float* h0   = (const float*)d_in[2];
    const float* c0   = (const float*)d_in[3];
    const float* w_ih = (const float*)d_in[4];
    const float* w_hh = (const float*)d_in[5];
    const float* b_ih = (const float*)d_in[6];
    const float* b_hh = (const float*)d_in[7];
    const float* w_se = (const float*)d_in[8];
    const float* b_se = (const float*)d_in[9];
    const float* w_hp = (const float*)d_in[10];
    const float* b_hp = (const float*)d_in[11];
    float* out = (float*)d_out;

    dim3 grid(NPEDS / PEDS_PER_BLOCK);  // 512 blocks x 256 threads (8 waves)
    decoder_lstm_wmma<<<grid, TPB, SMEM_BYTES, stream>>>(
        last_pos_rel, h0, c0, w_ih, w_hh, b_ih, b_hh, w_se, b_se, w_hp, b_hp, out);
}
